// GAT_7842610282842
// MI455X (gfx1250) — matrix operand order repaired and checked by reading, not compiled, not run on hardware
//
#include <hip/hip_runtime.h>
#include <hip/hip_bf16.h>

#define NNODES 50000
#define FIN    128
#define HIDC   64
#define NHEADS 4
#define NEDGES 800000
#define NGRAPH 64

typedef __attribute__((ext_vector_type(16))) __bf16 v16bf;
typedef __attribute__((ext_vector_type(8)))  __bf16 v8bf;
typedef __attribute__((ext_vector_type(8)))  float  v8f;

__device__ __forceinline__ __bf16 to_bf16(float f) {
    unsigned u = __float_as_uint(f);
    unsigned r = u + 0x7FFFu + ((u >> 16) & 1u);   // round-to-nearest-even
    unsigned short hs = (unsigned short)(r >> 16);
    __bf16 b;
    __builtin_memcpy(&b, &hs, sizeof(b));
    return b;
}

// ---------- generic helpers ----------
__global__ void k_fill_f32(float* p, float v, long long n) {
    long long i = (long long)blockIdx.x * blockDim.x + threadIdx.x;
    if (i < n) p[i] = v;
}

__global__ void k_f32_to_bf16(const float* __restrict__ in, __bf16* __restrict__ out, long long n) {
    long long i = (long long)blockIdx.x * blockDim.x + threadIdx.x;
    if (i < n) out[i] = to_bf16(in[i]);
}

// W[K,Ncol] f32 (row-major) -> Wp[Ncol,K] bf16  (B operand packed per-column)
__global__ void k_transpose_bf16(const float* __restrict__ W, __bf16* __restrict__ Wp,
                                 int K, int Ncol) {
    int idx = blockIdx.x * blockDim.x + threadIdx.x;
    if (idx >= K * Ncol) return;
    int k = idx / Ncol, n = idx % Ncol;
    Wp[(long long)n * K + k] = to_bf16(W[idx]);
}

// ---------- WMMA GEMM: C[M,Ncols] f32 = A[M,K] bf16 @ Bp[Ncols,K] bf16 ----------
// One wave computes one 16x16 tile; K-loop in steps of 32 with v_wmma_f32_16x16x32_bf16.
__global__ void k_wmma_gemm_bf16(const __bf16* __restrict__ A,
                                 const __bf16* __restrict__ Bp,
                                 float* __restrict__ C,
                                 int M, int Ncols, int K) {
    int wave = (int)((blockIdx.x * blockDim.x + threadIdx.x) >> 5);
    int lane = threadIdx.x & 31;
    int ntiles = Ncols >> 4;
    int tileM  = wave / ntiles;
    int tileN  = wave - tileM * ntiles;
    if (tileM * 16 >= M) return;

    const int half = lane >> 4;      // 0: lanes 0-15, 1: lanes 16-31
    const int l15  = lane & 15;
    const int rowA = tileM * 16 + l15;
    const int colB = tileN * 16 + l15;

    v8f acc = {};
    for (int k0 = 0; k0 < K; k0 += 32) {
        // A fragment (ISA 16-bit A layout): lane half 0 holds K {0..7,16..23}, half 1 {8..15,24..31}
        const __bf16* ap = A + (long long)rowA * K + k0 + half * 8;
        v8bf a0 = *(const v8bf*)(ap);
        v8bf a1 = *(const v8bf*)(ap + 16);
        if (k0 + 32 < K) __builtin_prefetch(ap + 32, 0, 3);   // global_prefetch_b8
        v16bf af;
#pragma unroll
        for (int i = 0; i < 8; ++i) { af[i] = a0[i]; af[i + 8] = a1[i]; }

        // B fragment: lane holds column colB, same K layout as A: half 0 {0..7,16..23}, half 1 {8..15,24..31}
        const __bf16* bptr = Bp + (long long)colB * K + k0 + half * 8;
        v8bf b0 = *(const v8bf*)(bptr);
        v8bf b1 = *(const v8bf*)(bptr + 16);
        v16bf bfr;
#pragma unroll
        for (int i = 0; i < 8; ++i) { bfr[i] = b0[i]; bfr[i + 8] = b1[i]; }

        acc = __builtin_amdgcn_wmma_f32_16x16x32_bf16(
                  false, af, false, bfr, (short)0, acc, false, false);
    }
    // C/D f32 16x16 layout: lane n = l15, rows (half*8 .. half*8+7) in acc[0..7]
    float* cp = C + (long long)(tileM * 16 + half * 8) * Ncols + tileN * 16 + l15;
#pragma unroll
    for (int i = 0; i < 8; ++i) cp[(long long)i * Ncols] = acc[i];
}

// ---------- attention logits: al_s[i,h]=<h[i,h,:],a_src[h,:]>, same for dst ----------
__global__ void k_att_logits(const float* __restrict__ h,
                             const float* __restrict__ a_src,
                             const float* __restrict__ a_dst,
                             float* __restrict__ al_s, float* __restrict__ al_d,
                             int n, int H, int Cc) {
    int idx = blockIdx.x * blockDim.x + threadIdx.x;
    if (idx >= n * H) return;
    int i = idx / H, hd = idx - i * H;
    const float* hp = h + ((long long)i * H + hd) * Cc;
    const float* as = a_src + hd * Cc;
    const float* ad = a_dst + hd * Cc;
    float s = 0.f, d = 0.f;
#pragma unroll 4
    for (int c = 0; c < Cc; ++c) { float v = hp[c]; s += v * as[c]; d += v * ad[c]; }
    al_s[idx] = s; al_d[idx] = d;
}

// float atomic max via signed-max / unsigned-min punning (init with -inf)
__device__ __forceinline__ void atomicMaxFloat(float* addr, float val) {
    if (!(val == val)) return;
    if (val >= 0.f) atomicMax((int*)addr, __float_as_int(val));
    else            atomicMin((unsigned int*)addr, (unsigned int)__float_as_int(val));
}

// ---------- edge pass 1: segment max of leaky-relu logits ----------
__global__ void k_edge_max(const int* __restrict__ ei, int E_, int n,
                           const float* __restrict__ al_s, const float* __restrict__ al_d,
                           float* __restrict__ m, int H) {
    long long idx = (long long)blockIdx.x * blockDim.x + threadIdx.x;
    long long tot = (long long)(E_ + n) * H;
    if (idx >= tot) return;
    int hd = (int)(idx % H);
    int e  = (int)(idx / H);
    int s, d;
    if (e < E_) { s = ei[e]; d = ei[E_ + e]; } else { s = d = e - E_; }
    float logit = al_s[(long long)s * H + hd] + al_d[(long long)d * H + hd];
    logit = logit > 0.f ? logit : 0.2f * logit;
    atomicMaxFloat(&m[(long long)d * H + hd], logit);
}

__global__ void k_fix_m(float* m, long long n) {
    long long i = (long long)blockIdx.x * blockDim.x + threadIdx.x;
    if (i >= n) return;
    float v = m[i];
    if (!(v >= -3.0e38f && v <= 3.0e38f)) m[i] = 0.f;   // non-finite -> 0
}

// ---------- edge pass 2: fused exp + denom + unnormalized weighted scatter-sum ----------
__global__ void k_edge_acc(const int* __restrict__ ei, int E_, int n,
                           const float* __restrict__ al_s, const float* __restrict__ al_d,
                           const float* __restrict__ m,
                           const float* __restrict__ h,
                           float* __restrict__ acc, float* __restrict__ denom,
                           int H, int Cc) {
    long long idx = (long long)blockIdx.x * blockDim.x + threadIdx.x;
    long long tot = (long long)(E_ + n) * H * Cc;
    if (idx >= tot) return;
    int c = (int)(idx % Cc);
    long long t = idx / Cc;
    int hd = (int)(t % H);
    int e  = (int)(t / H);
    int s, d;
    if (e < E_) { s = ei[e]; d = ei[E_ + e]; } else { s = d = e - E_; }
    float logit = al_s[(long long)s * H + hd] + al_d[(long long)d * H + hd];
    logit = logit > 0.f ? logit : 0.2f * logit;
    float ex = __expf(logit - m[(long long)d * H + hd]);
    if (c == 0) atomicAdd(&denom[(long long)d * H + hd], ex);
    float v = h[((long long)s * H + hd) * Cc + c];
    atomicAdd(&acc[((long long)d * H + hd) * Cc + c], ex * v);
}

// ---------- layer-1 finalize: normalize, +bias, ELU, -> bf16 for next GEMM ----------
__global__ void k_finalize1(const float* __restrict__ acc, const float* __restrict__ denom,
                            const float* __restrict__ bias, __bf16* __restrict__ out_bf,
                            int n, int H, int Cc) {
    long long idx = (long long)blockIdx.x * blockDim.x + threadIdx.x;
    long long tot = (long long)n * H * Cc;
    if (idx >= tot) return;
    int r = (int)(idx % (H * Cc));
    long long i = idx / (H * Cc);
    int hd = r / Cc;
    float v = acc[idx] / (denom[i * H + hd] + 1e-16f) + bias[r];
    v = v > 0.f ? v : __expf(v) - 1.f;          // ELU
    out_bf[idx] = to_bf16(v);
}

// ---------- layer-2 finalize fused with mean-pool scatter ----------
__global__ void k_finalize2_pool(const float* __restrict__ acc, const float* __restrict__ denom,
                                 const float* __restrict__ bias, const int* __restrict__ batch,
                                 float* __restrict__ pooled, int n, int Cc) {
    long long idx = (long long)blockIdx.x * blockDim.x + threadIdx.x;
    long long tot = (long long)n * Cc;
    if (idx >= tot) return;
    int c = (int)(idx % Cc);
    long long i = idx / Cc;
    float v = acc[idx] / (denom[i] + 1e-16f) + bias[c];
    v = v > 0.f ? v : __expf(v) - 1.f;          // ELU
    atomicAdd(&pooled[(long long)batch[i] * Cc + c], v);
}

__global__ void k_count(const int* __restrict__ batch, float* __restrict__ cnt, int n) {
    int i = blockIdx.x * blockDim.x + threadIdx.x;
    if (i < n) atomicAdd(&cnt[batch[i]], 1.f);
}

// ---------- head: out[g] = <pooled[g]/max(cnt,1), fcW> + fcb ----------
__global__ void k_head(const float* __restrict__ pooled, const float* __restrict__ cnt,
                       const float* __restrict__ fcW, const float* __restrict__ fcb,
                       float* __restrict__ out, int ng, int Cc) {
    int g = blockIdx.x * blockDim.x + threadIdx.x;
    if (g >= ng) return;
    float cn = cnt[g] > 1.f ? cnt[g] : 1.f;
    float sum = 0.f;
    for (int c = 0; c < Cc; ++c) sum += (pooled[(long long)g * Cc + c] / cn) * fcW[c];
    out[g] = sum + fcb[0];
}

static inline unsigned gdim(long long n, int b) { return (unsigned)((n + b - 1) / b); }

extern "C" void kernel_launch(void* const* d_in, const int* in_sizes, int n_in,
                              void* d_out, int out_size, void* d_ws, size_t ws_size,
                              hipStream_t stream) {
    (void)in_sizes; (void)n_in; (void)out_size; (void)ws_size;
    const float* x     = (const float*)d_in[0];
    const int*   ei    = (const int*)  d_in[1];   // [2,E] flat
    const int*   batch = (const int*)  d_in[2];
    const float* W1    = (const float*)d_in[3];
    const float* as1   = (const float*)d_in[4];
    const float* ad1   = (const float*)d_in[5];
    const float* b1    = (const float*)d_in[6];
    const float* W2    = (const float*)d_in[7];
    const float* as2   = (const float*)d_in[8];
    const float* ad2   = (const float*)d_in[9];
    const float* b2    = (const float*)d_in[10];
    const float* fcW   = (const float*)d_in[11];
    const float* fcb   = (const float*)d_in[12];
    float* out = (float*)d_out;

    const int N = NNODES, H1 = NHEADS, C = HIDC, D1 = NHEADS * HIDC; // 256
    char* ws = (char*)d_ws;
    size_t off = 0;
    auto alloc = [&](size_t bytes) { size_t o = off; off += (bytes + 255) & ~(size_t)255; return o; };
    __bf16* xb   = (__bf16*)(ws + alloc((size_t)N * FIN * 2));
    __bf16* wb1  = (__bf16*)(ws + alloc((size_t)D1 * FIN * 2));   // [256,128]
    __bf16* wb2  = (__bf16*)(ws + alloc((size_t)C * D1 * 2));     // [64,256]
    float*  h1   = (float*) (ws + alloc((size_t)N * D1 * 4));
    float*  als1v= (float*) (ws + alloc((size_t)N * H1 * 4));
    float*  ald1v= (float*) (ws + alloc((size_t)N * H1 * 4));
    float*  m1   = (float*) (ws + alloc((size_t)N * H1 * 4));
    float*  dn1  = (float*) (ws + alloc((size_t)N * H1 * 4));
    float*  acc1 = (float*) (ws + alloc((size_t)N * D1 * 4));
    __bf16* h1b  = (__bf16*)(ws + alloc((size_t)N * D1 * 2));
    float*  h2   = (float*) (ws + alloc((size_t)N * C * 4));
    float*  als2v= (float*) (ws + alloc((size_t)N * 4));
    float*  ald2v= (float*) (ws + alloc((size_t)N * 4));
    float*  m2   = (float*) (ws + alloc((size_t)N * 4));
    float*  dn2  = (float*) (ws + alloc((size_t)N * 4));
    float*  acc2 = (float*) (ws + alloc((size_t)N * C * 4));
    float*  pool = (float*) (ws + alloc((size_t)NGRAPH * C * 4));
    float*  cnt  = (float*) (ws + alloc((size_t)NGRAPH * 4));

    const float NEG_INF = -__builtin_inff();

    // --- prep: bf16 conversions / weight transposes ---
    k_f32_to_bf16<<<gdim((long long)N * FIN, 256), 256, 0, stream>>>(x, xb, (long long)N * FIN);
    k_transpose_bf16<<<gdim((long long)FIN * D1, 256), 256, 0, stream>>>(W1, wb1, FIN, D1);
    k_transpose_bf16<<<gdim((long long)D1 * C, 256), 256, 0, stream>>>(W2, wb2, D1, C);

    // --- layer 1 GEMM: h1 = x @ W1  (WMMA bf16, f32 accum) ---
    {
        long long waves = ((long long)N / 16) * (D1 / 16);
        k_wmma_gemm_bf16<<<gdim(waves * 32, 256), 256, 0, stream>>>(xb, wb1, h1, N, D1, FIN);
    }
    k_att_logits<<<gdim((long long)N * H1, 256), 256, 0, stream>>>(h1, as1, ad1, als1v, ald1v, N, H1, C);

    k_fill_f32<<<gdim((long long)N * H1, 256), 256, 0, stream>>>(m1, NEG_INF, (long long)N * H1);
    k_fill_f32<<<gdim((long long)N * H1, 256), 256, 0, stream>>>(dn1, 0.f, (long long)N * H1);
    k_fill_f32<<<gdim((long long)N * D1, 256), 256, 0, stream>>>(acc1, 0.f, (long long)N * D1);

    k_edge_max<<<gdim((long long)(NEDGES + N) * H1, 256), 256, 0, stream>>>(ei, NEDGES, N, als1v, ald1v, m1, H1);
    k_fix_m<<<gdim((long long)N * H1, 256), 256, 0, stream>>>(m1, (long long)N * H1);
    k_edge_acc<<<gdim((long long)(NEDGES + N) * H1 * C, 256), 256, 0, stream>>>(
        ei, NEDGES, N, als1v, ald1v, m1, h1, acc1, dn1, H1, C);
    k_finalize1<<<gdim((long long)N * D1, 256), 256, 0, stream>>>(acc1, dn1, b1, h1b, N, H1, C);

    // --- layer 2 GEMM: h2 = elu(out1) @ W2 ---
    {
        long long waves = ((long long)N / 16) * (C / 16);
        k_wmma_gemm_bf16<<<gdim(waves * 32, 256), 256, 0, stream>>>(h1b, wb2, h2, N, C, D1);
    }
    k_att_logits<<<gdim((long long)N, 256), 256, 0, stream>>>(h2, as2, ad2, als2v, ald2v, N, 1, C);

    k_fill_f32<<<gdim((long long)N, 256), 256, 0, stream>>>(m2, NEG_INF, (long long)N);
    k_fill_f32<<<gdim((long long)N, 256), 256, 0, stream>>>(dn2, 0.f, (long long)N);
    k_fill_f32<<<gdim((long long)N * C, 256), 256, 0, stream>>>(acc2, 0.f, (long long)N * C);
    k_fill_f32<<<gdim((long long)NGRAPH * C, 256), 256, 0, stream>>>(pool, 0.f, (long long)NGRAPH * C);
    k_fill_f32<<<gdim((long long)NGRAPH, 256), 256, 0, stream>>>(cnt, 0.f, (long long)NGRAPH);

    k_edge_max<<<gdim((long long)(NEDGES + N), 256), 256, 0, stream>>>(ei, NEDGES, N, als2v, ald2v, m2, 1);
    k_fix_m<<<gdim((long long)N, 256), 256, 0, stream>>>(m2, (long long)N);
    k_edge_acc<<<gdim((long long)(NEDGES + N) * C, 256), 256, 0, stream>>>(
        ei, NEDGES, N, als2v, ald2v, m2, h2, acc2, dn2, 1, C);

    k_finalize2_pool<<<gdim((long long)N * C, 256), 256, 0, stream>>>(acc2, dn2, b2, batch, pool, N, C);
    k_count<<<gdim((long long)N, 256), 256, 0, stream>>>(batch, cnt, N);

    k_head<<<1, 64, 0, stream>>>(pool, cnt, fcW, fcb, out, NGRAPH, C);
}
